// get_model_39986145526342
// MI455X (gfx1250) — matrix operand order repaired and checked by reading, not compiled, not run on hardware
//
#include <hip/hip_runtime.h>
#include <math.h>

// ---------------------------------------------------------------------------
// MambaIR-style x2 SR model on gfx1250 (MI455X).
// - All channel-mixing matmuls: v_wmma_f32_16x16x32_f16 (f16 in, f32 acc).
// - GEMM B panels staged in LDS via gfx1250 async-to-LDS DMA (ASYNCcnt).
// - Selective scan parallelized 32x with a chunked two-pass prefix scan.
// Layouts: activations are token-major HWC: X[pixel*C + c].
// ---------------------------------------------------------------------------

#define LEAK 0.1f
#define KBNS 0.9999950000374997f   // rsqrt(1 + 1e-5), folded BN scale

static __device__ __forceinline__ float lrelu_f(float v) { return v >= 0.f ? v : LEAK * v; }

typedef __attribute__((ext_vector_type(16))) _Float16 v16h;
typedef __attribute__((ext_vector_type(8)))  float    v8f;

static __device__ __forceinline__ v8f wmma_f16(v16h a, v16h b, v8f c) {
  // 8 args: (neg_a, A, neg_b, B, c_mod, C, reuse_a, reuse_b)
  return __builtin_amdgcn_wmma_f32_16x16x32_f16(false, a, false, b, (short)0, c, false, false);
}

// gfx1250 async global->LDS DMA (tracked by ASYNCcnt, not LOADcnt)
static __device__ __forceinline__ void async_load_b32(uint32_t lds_off, const float* gaddr) {
  asm volatile("global_load_async_to_lds_b32 %0, %1, off"
               :: "v"(lds_off), "v"(gaddr) : "memory");
}
static __device__ __forceinline__ void async_wait0() {
  asm volatile("s_wait_asynccnt 0" ::: "memory");
}

struct AConcat { const float* p[4]; };

// ---------------------------------------------------------------------------
// Generic token GEMM:  Out[M,N] = concat_k(A parts)[M,K] * B[K,N]
//   A: up to 4 parts, each 64 channels wide, row stride ldA (per part)
//   B: BT=1 -> B stored [N][K] (conv OIHW 1x1 weights); BT=0 -> B stored [K][N]
// One workgroup (8 waves) owns one 16-wide N-tile: the K x 16 B panel is
// DMA'd into LDS once (zero-padded past N), then each wave runs a branch-free
// WMMA K-loop over its own 16-token M-tile (8 M-tiles per workgroup).
// ---------------------------------------------------------------------------
#define GEMM_WAVES 8

__global__ void gemm_tok(AConcat A, int ldA, int nParts,
                         const float* __restrict__ B, int BT,
                         float* __restrict__ Out, int ldOut,
                         int M, int N,
                         const float* __restrict__ scale,
                         const float* __restrict__ shift,
                         const float* __restrict__ resid,
                         int act)
{
  __shared__ float sB[256 * 16];               // B panel [k][n], K <= 256
  const int K = nParts * 64;
  const int mTiles = M >> 4;
  const int nStrips = (mTiles + GEMM_WAVES - 1) / GEMM_WAVES;
  const int nt = blockIdx.x / nStrips;
  const int strip = blockIdx.x % nStrips;
  const int n0 = nt << 4;

  // ---- stage B panel into LDS with async-to-LDS DMA ----
  const uint32_t ldsBase = (uint32_t)(uintptr_t)&sB[0];
  for (int e = threadIdx.x; e < K * 16; e += blockDim.x) {
    const int k = e >> 4, n = e & 15;
    const int gn = n0 + n;
    if (gn < N) {
      const float* g = BT ? (B + (size_t)gn * K + k) : (B + (size_t)k * N + gn);
      async_load_b32(ldsBase + (uint32_t)e * 4u, g);
    } else {
      sB[e] = 0.f;                             // zero-pad ragged N tail
    }
  }
  async_wait0();
  __syncthreads();

  const int wv = threadIdx.x >> 5;
  const int mt = strip * GEMM_WAVES + wv;
  const int lane = threadIdx.x & 31;
  const int lh = lane >> 4;                    // lane half (wave32)
  const int ml = lane & 15;
  const int bcol = n0 + ml;

  if (mt < mTiles) {                           // wave-uniform: EXEC all-1s
    const int arow = (mt << 4) + ml;
    v8f acc = {0.f, 0.f, 0.f, 0.f, 0.f, 0.f, 0.f, 0.f};
#pragma unroll 2
    for (int ks = 0; ks < K; ks += 32) {
      const float* Ap = A.p[ks >> 6] + (size_t)arow * ldA + (ks & 63);
      __builtin_prefetch(Ap + ldA, 0, 1);      // global_prefetch_b8
      v16h av, bv;
      // ISA 16-bit A 16x32 layout: lanes0-15 K={0..7,16..23}, lanes16-31 +8
#pragma unroll
      for (int j = 0; j < 8; ++j) {
        const int k0 = ((j & 3) << 1) + ((j >= 4) ? 16 : 0) + (lh ? 8 : 0);
        av[2 * j]     = (_Float16)Ap[k0];
        av[2 * j + 1] = (_Float16)Ap[k0 + 1];
      }
      // B 32x16 layout matches A: lanes0-15 K={0..7,16..23}, lanes16-31 +8
#pragma unroll
      for (int j = 0; j < 8; ++j) {
        const int kk = ks + ((j & 3) << 1) + ((j >= 4) ? 16 : 0) + (lh ? 8 : 0);
        bv[2 * j]     = (_Float16)sB[kk * 16 + ml];
        bv[2 * j + 1] = (_Float16)sB[kk * 16 + 16 + ml];
      }
      acc = wmma_f16(av, bv, acc);
    }
    if (bcol < N) {
#pragma unroll
      for (int r = 0; r < 8; ++r) {
        const int row = (mt << 4) + r + (lh ? 8 : 0);
        float v = acc[r];
        if (scale) v = v * (scale[bcol] * KBNS) + shift[bcol];
        if (act)   v = lrelu_f(v);
        if (resid) v += resid[(size_t)row * ldOut + bcol];
        Out[(size_t)row * ldOut + bcol] = v;
      }
    }
  }
}

// ---------------------------------------------------------------------------
// Dense 3x3 conv as implicit GEMM with WMMA (pad=1, HWC layout).
// One wave: 16 consecutive pixels (one image row since W%16==0) x 16 out-ch.
// ---------------------------------------------------------------------------
__global__ void conv3x3_wmma(const float* __restrict__ X,
                             const float* __restrict__ Wt,   // (Co,Ci,3,3)
                             float* __restrict__ Out,
                             int Himg, int Wimg, int Ci, int Co, int act)
{
  const int M = Himg * Wimg;
  const int mTiles = M >> 4;
  const int nTiles = Co >> 4;
  const int w = blockIdx.x * (blockDim.x >> 5) + (threadIdx.x >> 5);
  if (w >= mTiles * nTiles) return;
  const int mt = w % mTiles, nt = w / mTiles;
  const int lane = threadIdx.x & 31;
  const int lh = lane >> 4, ml = lane & 15;
  const int p0 = mt << 4;
  const int y = p0 / Wimg, x0 = p0 % Wimg;
  const int bcol = (nt << 4) + ml;

  v8f acc = {0.f, 0.f, 0.f, 0.f, 0.f, 0.f, 0.f, 0.f};
  for (int ky = -1; ky <= 1; ++ky) {
    const int yy = y + ky;
    const bool rok = (yy >= 0 && yy < Himg);
    for (int kx = -1; kx <= 1; ++kx) {
      const int xx = x0 + ml + kx;
      const bool ok = rok && (xx >= 0) && (xx < Wimg);
      const float* Ap = X + ((size_t)yy * Wimg + xx) * Ci;
      const int tap = (ky + 1) * 3 + (kx + 1);
      for (int ks = 0; ks < Ci; ks += 32) {
        v16h av, bv;
#pragma unroll
        for (int j = 0; j < 8; ++j) {
          const int k0 = ((j & 3) << 1) + ((j >= 4) ? 16 : 0) + (lh ? 8 : 0) + ks;
          av[2 * j]     = ok ? (_Float16)Ap[k0]     : (_Float16)0.f;
          av[2 * j + 1] = ok ? (_Float16)Ap[k0 + 1] : (_Float16)0.f;
        }
#pragma unroll
        for (int j = 0; j < 8; ++j) {
          const int kg = ks + ((j & 3) << 1) + ((j >= 4) ? 16 : 0) + (lh ? 8 : 0);
          const float* s = Wt + ((size_t)bcol * Ci + kg) * 9 + tap;
          bv[2 * j]     = (_Float16)s[0];
          bv[2 * j + 1] = (_Float16)s[9];
        }
        acc = wmma_f16(av, bv, acc);
      }
    }
  }
#pragma unroll
  for (int r = 0; r < 8; ++r) {
    const int row = p0 + r + (lh ? 8 : 0);
    float v = acc[r];
    if (act) v = lrelu_f(v);
    Out[(size_t)row * Co + bcol] = v;
  }
}

// ---------------------------------------------------------------------------
// Elementwise / small kernels
// ---------------------------------------------------------------------------
static __device__ __forceinline__ float cubw(float s) {  // Keys cubic, a=-0.5
  s = fabsf(s);
  const float a = -0.5f;
  if (s <= 1.f) return ((a + 2.f) * s - (a + 3.f)) * s * s + 1.f;
  if (s < 2.f)  return (((s - 5.f) * s + 8.f) * s - 4.f) * a;
  return 0.f;
}

__global__ void bicubic_k(const float* __restrict__ xin, float* __restrict__ Out,
                          int Hi, int Wi, int Ho, int Wo)
{
  const int p = blockIdx.x * blockDim.x + threadIdx.x;
  if (p >= Ho * Wo) return;
  const int oy = p / Wo, ox = p % Wo;
  const float fy = (oy + 0.5f) * ((float)Hi / (float)Ho) - 0.5f;
  const float fx = (ox + 0.5f) * ((float)Wi / (float)Wo) - 0.5f;
  const int iy = (int)floorf(fy), ix = (int)floorf(fx);
  const float ty = fy - iy, tx = fx - ix;
  float wy[4], wx[4];
#pragma unroll
  for (int j = 0; j < 4; ++j) { wy[j] = cubw(ty - (j - 1)); wx[j] = cubw(tx - (j - 1)); }
  float acc = 0.f;
#pragma unroll
  for (int jy = 0; jy < 4; ++jy) {
    int yy = iy + jy - 1; yy = yy < 0 ? 0 : (yy >= Hi ? Hi - 1 : yy);
#pragma unroll
    for (int jx = 0; jx < 4; ++jx) {
      int xx = ix + jx - 1; xx = xx < 0 ? 0 : (xx >= Wi ? Wi - 1 : xx);
      acc += wy[jy] * wx[jx] * xin[yy * Wi + xx];
    }
  }
  Out[p] = acc;
}

__global__ void shallow_k(const float* __restrict__ xin, const float* __restrict__ wsh,
                          float* __restrict__ Out, int Himg, int Wimg)
{
  const int idx = blockIdx.x * blockDim.x + threadIdx.x;
  if (idx >= Himg * Wimg * 64) return;
  const int c = idx & 63, p = idx >> 6;
  const int y = p / Wimg, x = p % Wimg;
  float acc = 0.f;
  for (int ky = -1; ky <= 1; ++ky)
    for (int kx = -1; kx <= 1; ++kx) {
      const int yy = y + ky, xx = x + kx;
      if (yy < 0 || yy >= Himg || xx < 0 || xx >= Wimg) continue;
      acc += xin[yy * Wimg + xx] * wsh[c * 9 + (ky + 1) * 3 + (kx + 1)];
    }
  Out[idx] = lrelu_f(acc);
}

// LPE depthwise 3x3 + BN + lrelu
__global__ void dw3_bn_k(const float* __restrict__ X, const float* __restrict__ wdw,
                         const float* __restrict__ g, const float* __restrict__ b,
                         float* __restrict__ Out, int Himg, int Wimg)
{
  const int idx = blockIdx.x * blockDim.x + threadIdx.x;
  if (idx >= Himg * Wimg * 64) return;
  const int c = idx & 63, p = idx >> 6;
  const int y = p / Wimg, x = p % Wimg;
  float acc = 0.f;
  for (int ky = -1; ky <= 1; ++ky)
    for (int kx = -1; kx <= 1; ++kx) {
      const int yy = y + ky, xx = x + kx;
      if (yy < 0 || yy >= Himg || xx < 0 || xx >= Wimg) continue;
      acc += X[((size_t)yy * Wimg + xx) * 64 + c] * wdw[c * 9 + (ky + 1) * 3 + (kx + 1)];
    }
  acc = acc * (g[c] * KBNS) + b[c];
  Out[idx] = lrelu_f(acc);
}

// Multi-scale block: group0 = dense 1x1 (16->16), groups1..3 = depthwise 3/5/7
__global__ void ms_multiscale_k(const float* __restrict__ X,
                                const float* __restrict__ w1, const float* __restrict__ w3,
                                const float* __restrict__ w5, const float* __restrict__ w7,
                                float* __restrict__ Out, int Himg, int Wimg)
{
  const int idx = blockIdx.x * blockDim.x + threadIdx.x;
  if (idx >= Himg * Wimg * 64) return;
  const int c = idx & 63, p = idx >> 6;
  const int y = p / Wimg, x = p % Wimg;
  const int g = c >> 4, cg = c & 15;
  float acc = 0.f;
  if (g == 0) {
    const float* xr = X + (size_t)p * 64;
#pragma unroll
    for (int ci = 0; ci < 16; ++ci) acc += w1[cg * 16 + ci] * xr[ci];
  } else {
    const int rad = g;                  // kernel sizes 3,5,7 -> radius 1,2,3
    const int kk = 2 * g + 1;
    const float* wp = (g == 1) ? (w3 + cg * 9) : ((g == 2) ? (w5 + cg * 25) : (w7 + cg * 49));
    for (int dy = -rad; dy <= rad; ++dy) {
      const int yy = y + dy;
      if (yy < 0 || yy >= Himg) continue;
      for (int dx = -rad; dx <= rad; ++dx) {
        const int xx = x + dx;
        if (xx < 0 || xx >= Wimg) continue;
        acc += X[((size_t)yy * Wimg + xx) * 64 + c] * wp[(dy + rad) * kk + (dx + rad)];
      }
    }
  }
  Out[idx] = acc;
}

__global__ void layernorm_k(const float* __restrict__ X, const float* __restrict__ g,
                            const float* __restrict__ b, float* __restrict__ Out, int L)
{
  const int p = blockIdx.x * blockDim.x + threadIdx.x;
  if (p >= L) return;
  const float* xr = X + (size_t)p * 64;
  float mu = 0.f;
#pragma unroll
  for (int c = 0; c < 64; ++c) mu += xr[c];
  mu *= (1.f / 64.f);
  float var = 0.f;
#pragma unroll
  for (int c = 0; c < 64; ++c) { const float d = xr[c] - mu; var += d * d; }
  var *= (1.f / 64.f);
  const float inv = rsqrtf(var + 1e-5f);
  float* orow = Out + (size_t)p * 64;
#pragma unroll
  for (int c = 0; c < 64; ++c) orow[c] = (xr[c] - mu) * inv * g[c] + b[c];
}

// ---------------------------------------------------------------------------
// 4-direction selective scan, chunked two-pass prefix scan (NCH chunks).
// Thread (c,d) owns one linear recurrence h' = ab*h + u; recurrences compose,
// so: pass1 = per-chunk (prod ab, h_end) with h0=0; mid = chain 32 chunk
// states; pass2 = rescan from correct h_init and emit outputs (d-reduction
// via wave32 16-lane butterfly shuffles, writes through dir permutation).
// ---------------------------------------------------------------------------
#define NCH 32

static __device__ __forceinline__ int dir_pos(int dir, int t, int L, int Himg, int Wimg) {
  const int s = (dir & 1) ? (L - 1 - t) : t;
  if (dir < 2) return s;
  const int col = s / Himg, row = s % Himg;
  return row * Wimg + col;
}

__global__ void ssm_scan_pass1(const float* __restrict__ xz, const float* __restrict__ xp,
                               const float* __restrict__ A_log,
                               float* __restrict__ chA, float* __restrict__ chH,
                               int Himg, int Wimg)
{
  const int dir = blockIdx.x / NCH, ch = blockIdx.x % NCH;
  const int tid = threadIdx.x;                  // 1024 = 64 ch x 16 states
  const int c = tid >> 4, d = tid & 15;
  const int L = Himg * Wimg, CL = L / NCH;
  const float a = -expf(A_log[c * 16 + d]);
  float h = 0.f, pA = 1.f;
  for (int t = ch * CL; t < (ch + 1) * CL; ++t) {
    const int pos = dir_pos(dir, t, L, Himg, Wimg);
    const float dr  = xp[(size_t)pos * 33];
    const float dlt = dr > 20.f ? dr : log1pf(expf(dr));      // softplus
    const float Bi  = xp[(size_t)pos * 33 + 1 + d];
    const float xv  = xz[(size_t)pos * 128 + c];
    const float ab  = fmaf(dlt, a, 1.f);
    h = fmaf(ab, h, dlt * Bi * xv);
    pA *= ab;
  }
  const size_t o = ((size_t)dir * NCH + ch) * 1024 + tid;
  chA[o] = pA;
  chH[o] = h;
}

__global__ void ssm_scan_mid(const float* __restrict__ chA, const float* __restrict__ chH,
                             float* __restrict__ hinit)
{
  const int dir = blockIdx.x;
  const int tid = threadIdx.x;
  float h = 0.f;
  for (int ch = 0; ch < NCH; ++ch) {
    const size_t o = ((size_t)dir * NCH + ch) * 1024 + tid;
    hinit[o] = h;
    h = fmaf(chA[o], h, chH[o]);
  }
}

__global__ void ssm_scan_pass2(const float* __restrict__ xz, const float* __restrict__ xp,
                               const float* __restrict__ A_log,
                               const float* __restrict__ hinit,
                               float* __restrict__ yd,        // (4,L,64)
                               int Himg, int Wimg)
{
  const int dir = blockIdx.x / NCH, ch = blockIdx.x % NCH;
  const int tid = threadIdx.x;
  const int c = tid >> 4, d = tid & 15;
  const int L = Himg * Wimg, CL = L / NCH;
  const float a = -expf(A_log[c * 16 + d]);
  float h = hinit[((size_t)dir * NCH + ch) * 1024 + tid];
  float* ybase = yd + (size_t)dir * L * 64;
  for (int t = ch * CL; t < (ch + 1) * CL; ++t) {
    const int pos = dir_pos(dir, t, L, Himg, Wimg);
    const float dr  = xp[(size_t)pos * 33];
    const float dlt = dr > 20.f ? dr : log1pf(expf(dr));
    const float Bi  = xp[(size_t)pos * 33 + 1 + d];
    const float Cc  = xp[(size_t)pos * 33 + 17 + d];
    const float xv  = xz[(size_t)pos * 128 + c];
    h = fmaf(fmaf(dlt, a, 1.f), h, dlt * Bi * xv);
    float contrib = h * Cc;
#pragma unroll
    for (int off = 8; off; off >>= 1) contrib += __shfl_xor(contrib, off, 16);
    if (d == 0) ybase[(size_t)pos * 64 + c] = contrib;
  }
}

__global__ void ssm_combine_k(const float* __restrict__ yd, const float* __restrict__ xz,
                              const float* __restrict__ Dp, float* __restrict__ gin, int L)
{
  const int idx = blockIdx.x * blockDim.x + threadIdx.x;
  if (idx >= L * 64) return;
  const int p = idx >> 6, c = idx & 63;
  const size_t n = (size_t)L * 64;
  const float y = 0.25f * (yd[idx] + yd[idx + n] + yd[idx + 2 * n] + yd[idx + 3 * n])
                + Dp[c] * xz[(size_t)p * 128 + c];
  const float z = xz[(size_t)p * 128 + 64 + c];
  gin[idx] = y * (z / (1.f + expf(-z)));                       // y * silu(z)
}

__global__ void pool_k(const float* __restrict__ X, float* __restrict__ pooled, int L)
{
  __shared__ float red[256];
  const int c = blockIdx.x;
  float s = 0.f;
  for (int p = threadIdx.x; p < L; p += blockDim.x) s += X[(size_t)p * 64 + c];
  red[threadIdx.x] = s;
  __syncthreads();
  for (int st = 128; st > 0; st >>= 1) {
    if ((int)threadIdx.x < st) red[threadIdx.x] += red[threadIdx.x + st];
    __syncthreads();
  }
  if (threadIdx.x == 0) pooled[c] = red[0] / (float)L;
}

__global__ void att_k(const float* __restrict__ pooled,
                      const float* __restrict__ c1w, const float* __restrict__ c1b,
                      const float* __restrict__ c2w, const float* __restrict__ c2b,
                      float* __restrict__ att)
{
  __shared__ float sp[64], sa[16];
  const int t = threadIdx.x;
  sp[t] = pooled[t];
  __syncthreads();
  if (t < 16) {
    float s = c1b[t];
    for (int c = 0; c < 64; ++c) s += c1w[t * 64 + c] * sp[c];
    sa[t] = fmaxf(s, 0.f);
  }
  __syncthreads();
  float s = c2b[t];
#pragma unroll
  for (int j = 0; j < 16; ++j) s += c2w[t * 16 + j] * sa[j];
  att[t] = 1.f / (1.f + expf(-s));
}

__global__ void att_apply_k(const float* __restrict__ fused, const float* __restrict__ att,
                            const float* __restrict__ fin, float* __restrict__ out, int n)
{
  const int idx = blockIdx.x * blockDim.x + threadIdx.x;
  if (idx >= n) return;
  out[idx] = fused[idx] * att[idx & 63] + fin[idx];
}

__global__ void shuffle_k(const float* __restrict__ Up, float* __restrict__ Out, int Ho, int Wo)
{
  const int idx = blockIdx.x * blockDim.x + threadIdx.x;
  if (idx >= Ho * Wo * 64) return;
  const int co = idx & 63, pp = idx >> 6;
  const int oy = pp / Wo, ox = pp % Wo;
  const int y = oy >> 1, ry = oy & 1, x = ox >> 1, rx = ox & 1;
  const float v = Up[((size_t)y * (Wo >> 1) + x) * 256 + co * 4 + ry * 2 + rx];
  Out[idx] = lrelu_f(v);
}

__global__ void head_k(const float* __restrict__ X, const float* __restrict__ wh,
                       const float* __restrict__ hb, const float* __restrict__ xup,
                       float* __restrict__ Out, int Himg, int Wimg)
{
  const int p = blockIdx.x * blockDim.x + threadIdx.x;
  if (p >= Himg * Wimg) return;
  const int y = p / Wimg, x = p % Wimg;
  float acc = hb[0] + xup[p];
  for (int ky = -1; ky <= 1; ++ky) {
    const int yy = y + ky;
    if (yy < 0 || yy >= Himg) continue;
    for (int kx = -1; kx <= 1; ++kx) {
      const int xx = x + kx;
      if (xx < 0 || xx >= Wimg) continue;
      const float* xr = X + ((size_t)yy * Wimg + xx) * 64;
      const int tap = (ky + 1) * 3 + (kx + 1);
      for (int c = 0; c < 64; ++c) acc += xr[c] * wh[c * 9 + tap];
    }
  }
  Out[p] = acc;
}

// ---------------------------------------------------------------------------
// Host orchestration
// ---------------------------------------------------------------------------
#define CDIV(a, b) (((a) + (b) - 1) / (b))

static inline void launch_gemm(const float* a0, const float* a1, const float* a2, const float* a3,
                               int nParts, int ldA, const float* B, int BT,
                               float* Out, int ldOut, int M, int N,
                               const float* scale, const float* shift, const float* resid,
                               int act, hipStream_t s)
{
  AConcat A; A.p[0] = a0; A.p[1] = a1 ? a1 : a0; A.p[2] = a2 ? a2 : a0; A.p[3] = a3 ? a3 : a0;
  const int mTiles = M / 16;
  const int nTiles = (N + 15) / 16;
  const int nStrips = CDIV(mTiles, GEMM_WAVES);
  gemm_tok<<<nTiles * nStrips, 256, 0, s>>>(A, ldA, nParts, B, BT, Out, ldOut, M, N,
                                            scale, shift, resid, act);
}

extern "C" void kernel_launch(void* const* d_in, const int* in_sizes, int n_in,
                              void* d_out, int out_size, void* d_ws, size_t ws_size,
                              hipStream_t stream)
{
  (void)in_sizes; (void)n_in; (void)out_size; (void)ws_size;
  const int H0 = 80, W0 = 80, L = H0 * W0;        // 6400 tokens
  const int Ho = 160, Wo = 160;

  const float* x      = (const float*)d_in[0];
  const float* sh_w   = (const float*)d_in[1];
  const float* lpe_dw = (const float*)d_in[2];
  const float* lpe_g  = (const float*)d_in[3];
  const float* lpe_b  = (const float*)d_in[4];
  const float* lpe_pw = (const float*)d_in[5];
  const float* ms_w1  = (const float*)d_in[6];
  const float* ms_w3  = (const float*)d_in[7];
  const float* ms_w5  = (const float*)d_in[8];
  const float* ms_w7  = (const float*)d_in[9];
  const float* ms_pw  = (const float*)d_in[10];
  const float* ms_g   = (const float*)d_in[11];
  const float* ms_b   = (const float*)d_in[12];
  const float* ln_g   = (const float*)d_in[13];
  const float* ln_b   = (const float*)d_in[14];
  const float* in_w   = (const float*)d_in[15];
  const float* A_log  = (const float*)d_in[16];
  const float* ssm_D  = (const float*)d_in[17];
  const float* xp_w   = (const float*)d_in[18];
  const float* out_w  = (const float*)d_in[19];
  const float* fuse_w = (const float*)d_in[20];
  const float* ca1_w  = (const float*)d_in[21];
  const float* ca1_b  = (const float*)d_in[22];
  const float* ca2_w  = (const float*)d_in[23];
  const float* ca2_b  = (const float*)d_in[24];
  const float* fe_w   = (const float*)d_in[25];
  const float* fl_w   = (const float*)d_in[26];
  const float* ff_w   = (const float*)d_in[27];
  const float* ref_w  = (const float*)d_in[28];
  const float* up_w   = (const float*)d_in[29];
  const float* head_w = (const float*)d_in[30];
  const float* head_b = (const float*)d_in[31];
  float* out = (float*)d_out;

  // Workspace carving (floats)
  float* wsf = (float*)d_ws;
  size_t o = 0;
  auto carve = [&](size_t n) { float* r = wsf + o; o += n; return r; };
  float* XUP   = carve((size_t)Ho * Wo);
  float* FEAT0 = carve((size_t)L * 64);    // shallow feature (kept for final skip)
  float* DW    = carve((size_t)L * 64);    // depthwise / multiscale temp
  float* T0    = carve((size_t)L * 64);    // f_local
  float* LN    = carve((size_t)L * 64);
  float* XZ    = carve((size_t)L * 128);   // [x_ssm | z]
  float* XP    = carve((size_t)L * 33);    // [dlt | B | C]
  float* YD    = carve((size_t)4 * L * 64);
  float* GIN   = carve((size_t)L * 64);    // y * silu(z)
  float* FG    = carve((size_t)L * 64);    // f_global
  float* FUSED = carve((size_t)L * 64);
  float* YS    = carve((size_t)8 * L * 64);
  float* EARLY = carve((size_t)L * 64);
  float* LATE  = carve((size_t)L * 64);
  float* FF    = carve((size_t)L * 64);
  float* REF   = carve((size_t)L * 64);
  float* UP    = carve((size_t)L * 256);
  float* SHUF  = carve((size_t)Ho * Wo * 64);
  float* POOLED= carve(64);
  float* ATT   = carve(64);
  float* CHA   = carve((size_t)4 * NCH * 1024);  // chunk prod(ab)
  float* CHH   = carve((size_t)4 * NCH * 1024);  // chunk local h_end
  float* HINIT = carve((size_t)4 * NCH * 1024);  // chunk initial h

  const int EW = CDIV(L * 64, 256);        // elementwise grid over (L,64)

  // Bicubic skip + shallow feature + LPE
  bicubic_k<<<CDIV(Ho * Wo, 256), 256, 0, stream>>>(x, XUP, H0, W0, Ho, Wo);
  shallow_k<<<EW, 256, 0, stream>>>(x, sh_w, FEAT0, H0, W0);
  dw3_bn_k<<<EW, 256, 0, stream>>>(FEAT0, lpe_dw, lpe_g, lpe_b, DW, H0, W0);
  launch_gemm(DW, 0, 0, 0, 1, 64, lpe_pw, 1, FEAT0, 64, L, 64, 0, 0, FEAT0, 0, stream);

  // 8 residual SSM blocks
  for (int b = 0; b < 8; ++b) {
    const float* fin = (b == 0) ? FEAT0 : (YS + (size_t)(b - 1) * L * 64);
    // local branch: multiscale conv -> 1x1 + BN + lrelu + residual
    ms_multiscale_k<<<EW, 256, 0, stream>>>(fin, ms_w1 + b * 256, ms_w3 + b * 144,
                                            ms_w5 + b * 400, ms_w7 + b * 784, DW, H0, W0);
    launch_gemm(DW, 0, 0, 0, 1, 64, ms_pw + b * 4096, 1, T0, 64, L, 64,
                ms_g + b * 64, ms_b + b * 64, fin, 1, stream);
    // global branch: LN -> in-proj -> x-proj -> 4-dir chunked scan -> gate -> out-proj
    layernorm_k<<<CDIV(L, 256), 256, 0, stream>>>(fin, ln_g + b * 64, ln_b + b * 64, LN, L);
    launch_gemm(LN, 0, 0, 0, 1, 64, in_w + b * 8192, 0, XZ, 128, L, 128, 0, 0, 0, 0, stream);
    launch_gemm(XZ, 0, 0, 0, 1, 128, xp_w + b * 2112, 0, XP, 33, L, 33, 0, 0, 0, 0, stream);
    ssm_scan_pass1<<<4 * NCH, 1024, 0, stream>>>(XZ, XP, A_log + b * 1024, CHA, CHH, H0, W0);
    ssm_scan_mid<<<4, 1024, 0, stream>>>(CHA, CHH, HINIT);
    ssm_scan_pass2<<<4 * NCH, 1024, 0, stream>>>(XZ, XP, A_log + b * 1024, HINIT, YD, H0, W0);
    ssm_combine_k<<<EW, 256, 0, stream>>>(YD, XZ, ssm_D + b * 64, GIN, L);
    launch_gemm(GIN, 0, 0, 0, 1, 64, out_w + b * 4096, 0, FG, 64, L, 64, 0, 0, 0, 0, stream);
    // fuse + channel attention + residual
    launch_gemm(T0, FG, 0, 0, 2, 64, fuse_w + b * 8192, 1, FUSED, 64, L, 64, 0, 0, 0, 0, stream);
    pool_k<<<64, 256, 0, stream>>>(FUSED, POOLED, L);
    att_k<<<1, 64, 0, stream>>>(POOLED, ca1_w + b * 1024, ca1_b + b * 16,
                                ca2_w + b * 1024, ca2_b + b * 64, ATT);
    att_apply_k<<<EW, 256, 0, stream>>>(FUSED, ATT, fin, YS + (size_t)b * L * 64, L * 64);
  }

  // hierarchical fusion
  launch_gemm(YS, YS + (size_t)L * 64, YS + (size_t)2 * L * 64, YS + (size_t)3 * L * 64,
              4, 64, fe_w, 1, EARLY, 64, L, 64, 0, 0, 0, 0, stream);
  launch_gemm(YS + (size_t)4 * L * 64, YS + (size_t)5 * L * 64, YS + (size_t)6 * L * 64,
              YS + (size_t)7 * L * 64, 4, 64, fl_w, 1, LATE, 64, L, 64, 0, 0, 0, 0, stream);
  launch_gemm(EARLY, LATE, 0, 0, 2, 64, ff_w, 1, FF, 64, L, 64, 0, 0, FEAT0, 0, stream);

  // refine, upsample, head
  conv3x3_wmma<<<CDIV((L / 16) * (64 / 16), 8), 256, 0, stream>>>(FF, ref_w, REF, H0, W0, 64, 64, 1);
  conv3x3_wmma<<<CDIV((L / 16) * (256 / 16), 8), 256, 0, stream>>>(REF, up_w, UP, H0, W0, 64, 256, 0);
  shuffle_k<<<CDIV(Ho * Wo * 64, 256), 256, 0, stream>>>(UP, SHUF, Ho, Wo);
  head_k<<<CDIV(Ho * Wo, 256), 256, 0, stream>>>(SHUF, head_w, head_b, XUP, out, Ho, Wo);
}
